// PointerAttention_55087250539005
// MI455X (gfx1250) — matrix operand order repaired and checked by reading, not compiled, not run on hardware
//
#include <hip/hip_runtime.h>
#include <hip/hip_bf16.h>

#define B_ 64
#define L_ 2048
#define D_ 512
#define H_ 512

typedef __attribute__((ext_vector_type(16))) __bf16 v16bf;
typedef __attribute__((ext_vector_type(8)))  __bf16 v8bf;
typedef __attribute__((ext_vector_type(8)))  float  v8f;

__device__ __forceinline__ __bf16 f2bf(float f) {
    union { float f; unsigned u; } v; v.f = f;
    unsigned r = v.u + 0x7FFFu + ((v.u >> 16) & 1u);   // round-to-nearest-even
    unsigned short h = (unsigned short)(r >> 16);
    return __builtin_bit_cast(__bf16, h);
}

// CDNA5 hardware tanh (trans op). Guarded: builtin if available, else raw ISA.
__device__ __forceinline__ float fast_tanh(float x) {
#if __has_builtin(__builtin_amdgcn_tanhf)
    return __builtin_amdgcn_tanhf(x);
#else
    float r;
    asm("v_tanh_f32 %0, %1\n\tv_nop" : "=v"(r) : "v"(x));  // v_nop: TRANS hazard rule
    return r;
#endif
}

// ---------- K0: convert Wc (fp32) -> bf16, pre-swizzled into WMMA A-fragment order.
// Fragment (htile,k) = 16x32 tile stored as 32 lanes x 16 bf16 (32B per lane,
// contiguous 1KB per fragment) matching the 16-bit A-matrix VGPR layout.
__global__ __launch_bounds__(256) void pa_cvt_wc(const float* __restrict__ Wc,
                                                 __bf16* __restrict__ out) {
    int i = blockIdx.x * 256 + threadIdx.x;          // H_*D_ total
    int h = i >> 9, d = i & (D_ - 1);
    int htile = h >> 4, m = h & 15;
    int k = d >> 5, kk = d & 31;
    int lane = m + 16 * ((kk >> 3) & 1);             // K {8-15,24-31} -> high lanes
    int j    = (kk & 7) + 8 * (kk >> 4);             // K {16..31} -> elements 8..15
    out[(((size_t)htile * 16 + k) * 32 + lane) * 16 + j] = f2bf(Wc[i]);
}

// ---------- K1: inp[b,h] = decoder_hidden[b,:] . Wi[h,:] + bi[h] ----------
__global__ __launch_bounds__(256) void pa_inp(const float* __restrict__ dh,
                                              const float* __restrict__ Wi,
                                              const float* __restrict__ bi,
                                              float* __restrict__ inp) {
    int idx = blockIdx.x * 256 + threadIdx.x;        // b*H + h
    int b = idx >> 9, h = idx & (H_ - 1);
    const float* x = dh + (size_t)b * D_;
    const float* w = Wi + (size_t)h * D_;
    float acc = bi[h];
#pragma unroll 8
    for (int d = 0; d < D_; ++d) acc += x[d] * w[d];
    inp[idx] = acc;
}

// ---------- K2: e[b,l] = sum_h V[h]*tanh(inp[b,h] + (Wc.ctx)[h,l] + bc[h]) ----------
// grid (L/64, B), block 256 = 8 waves. Workgroup tile: 64 l-columns x full H.
// Each wave owns 4 h-tiles, processed in TWO passes of 2 h-tiles x 4 l-tiles:
// 8 accumulator sets (64 VGPRs) per pass, one A fragment feeds 4 WMMAs (4x reuse),
// pass loop kept rolled + memory clobber so B fragments are not hoisted (no spills).
__global__ __launch_bounds__(256) void pa_energy(const float* __restrict__ ctxg,
                                                 const __bf16* __restrict__ wcs,
                                                 const float* __restrict__ inp,
                                                 const float* __restrict__ bc,
                                                 const float* __restrict__ V,
                                                 float* __restrict__ e_out) {
    __shared__ __bf16 sB[64][D_ + 8];   // 64 l-cols x 512 d (+8 pad) ~65KB
    __shared__ float  sInp[H_];
    __shared__ float  sBias[H_];
    __shared__ float  sV[H_];
    __shared__ float  sE[64];

    const int b   = blockIdx.y;
    const int l0  = blockIdx.x * 64;
    const int tid = threadIdx.x;

    // stage context tile (fp32 -> bf16), vectorized float4 reads
    const float* cbase = ctxg + ((size_t)b * L_ + l0) * D_;
    for (int i = tid; i < 64 * D_ / 4; i += 256) {
        int l = i >> 7, d4 = (i & 127) << 2;
        float4 v = *(const float4*)(cbase + (size_t)l * D_ + d4);
        sB[l][d4 + 0] = f2bf(v.x); sB[l][d4 + 1] = f2bf(v.y);
        sB[l][d4 + 2] = f2bf(v.z); sB[l][d4 + 3] = f2bf(v.w);
    }
    for (int i = tid; i < H_; i += 256) {
        sInp[i]  = inp[b * H_ + i];
        sBias[i] = bc[i];
        sV[i]    = V[i];
    }
    if (tid < 64) sE[tid] = 0.f;
    __syncthreads();

    const int lane = tid & 31;
    const int wave = tid >> 5;
    const int n    = lane & 15;
    const int half = lane >> 4;

    float esum[4] = {0.f, 0.f, 0.f, 0.f};

#pragma unroll 1
    for (int hg = 0; hg < 2; ++hg) {
        asm volatile("" ::: "memory");   // block LICM: B frags reload per pass
        const int ht0 = hg * 16 + wave;          // two h-tiles this pass
        const int ht1 = hg * 16 + 8 + wave;
        const v16bf* afr0 = (const v16bf*)wcs + (size_t)ht0 * 16 * 32 + lane;
        const v16bf* afr1 = (const v16bf*)wcs + (size_t)ht1 * 16 * 32 + lane;
        v8f c00 = {}, c01 = {}, c02 = {}, c03 = {};   // ht0 x lt0..3
        v8f c10 = {}, c11 = {}, c12 = {}, c13 = {};   // ht1 x lt0..3
#pragma unroll
        for (int k = 0; k < 16; ++k) {
            v16bf a0 = afr0[k * 32];
            v16bf a1 = afr1[k * 32];
            const int ko = k * 32 + half * 8;
            const __bf16* bp0 = &sB[n][ko];
            const __bf16* bp1 = &sB[16 + n][ko];
            const __bf16* bp2 = &sB[32 + n][ko];
            const __bf16* bp3 = &sB[48 + n][ko];
            v8bf x0 = *(const v8bf*)bp0, y0 = *(const v8bf*)(bp0 + 16);
            v8bf x1 = *(const v8bf*)bp1, y1 = *(const v8bf*)(bp1 + 16);
            v8bf x2 = *(const v8bf*)bp2, y2 = *(const v8bf*)(bp2 + 16);
            v8bf x3 = *(const v8bf*)bp3, y3 = *(const v8bf*)(bp3 + 16);
            v16bf b0 = __builtin_shufflevector(x0, y0, 0,1,2,3,4,5,6,7,8,9,10,11,12,13,14,15);
            v16bf b1 = __builtin_shufflevector(x1, y1, 0,1,2,3,4,5,6,7,8,9,10,11,12,13,14,15);
            v16bf b2 = __builtin_shufflevector(x2, y2, 0,1,2,3,4,5,6,7,8,9,10,11,12,13,14,15);
            v16bf b3 = __builtin_shufflevector(x3, y3, 0,1,2,3,4,5,6,7,8,9,10,11,12,13,14,15);
            c00 = __builtin_amdgcn_wmma_f32_16x16x32_bf16(false, a0, false, b0, (short)0, c00, false, false);
            c01 = __builtin_amdgcn_wmma_f32_16x16x32_bf16(false, a0, false, b1, (short)0, c01, false, false);
            c02 = __builtin_amdgcn_wmma_f32_16x16x32_bf16(false, a0, false, b2, (short)0, c02, false, false);
            c03 = __builtin_amdgcn_wmma_f32_16x16x32_bf16(false, a0, false, b3, (short)0, c03, false, false);
            c10 = __builtin_amdgcn_wmma_f32_16x16x32_bf16(false, a1, false, b0, (short)0, c10, false, false);
            c11 = __builtin_amdgcn_wmma_f32_16x16x32_bf16(false, a1, false, b1, (short)0, c11, false, false);
            c12 = __builtin_amdgcn_wmma_f32_16x16x32_bf16(false, a1, false, b2, (short)0, c12, false, false);
            c13 = __builtin_amdgcn_wmma_f32_16x16x32_bf16(false, a1, false, b3, (short)0, c13, false, false);
        }
        // C/D f32 layout: VGPR r -> M = half*8 + r, N = n
        const int hb0 = ht0 * 16 + half * 8;
        const int hb1 = ht1 * 16 + half * 8;
#pragma unroll
        for (int r = 0; r < 8; ++r) {
            const float pre0 = sInp[hb0 + r] + sBias[hb0 + r];
            const float vh0  = sV[hb0 + r];
            esum[0] += vh0 * fast_tanh(c00[r] + pre0);
            esum[1] += vh0 * fast_tanh(c01[r] + pre0);
            esum[2] += vh0 * fast_tanh(c02[r] + pre0);
            esum[3] += vh0 * fast_tanh(c03[r] + pre0);
            const float pre1 = sInp[hb1 + r] + sBias[hb1 + r];
            const float vh1  = sV[hb1 + r];
            esum[0] += vh1 * fast_tanh(c10[r] + pre1);
            esum[1] += vh1 * fast_tanh(c11[r] + pre1);
            esum[2] += vh1 * fast_tanh(c12[r] + pre1);
            esum[3] += vh1 * fast_tanh(c13[r] + pre1);
        }
    }
#pragma unroll
    for (int lt = 0; lt < 4; ++lt)
        esum[lt] += __shfl_xor(esum[lt], 16, 32);    // fold the two M-halves
    if (lane < 16) {
#pragma unroll
        for (int lt = 0; lt < 4; ++lt)
            atomicAdd(&sE[lt * 16 + n], esum[lt]);   // combine 8 waves
    }
    __syncthreads();
    if (tid < 64) e_out[(size_t)b * L_ + l0 + tid] = sE[tid];
}

// ---------- K3: masked softmax over L -> alpha ----------
__global__ __launch_bounds__(256) void pa_softmax(const float* __restrict__ e,
                                                  const unsigned char* __restrict__ mask,
                                                  float* __restrict__ alpha) {
    __shared__ float red[256];
    const int b = blockIdx.x, tid = threadIdx.x;
    float vals[8];
    float mx = -1e30f;
#pragma unroll
    for (int i = 0; i < 8; ++i) {
        int l = tid + i * 256;
        float v = e[(size_t)b * L_ + l];
        if (mask[(size_t)b * L_ + l]) v = -10.0f;   // reference: where(mask, -10, e)
        vals[i] = v;
        mx = fmaxf(mx, v);
    }
    red[tid] = mx; __syncthreads();
    for (int s = 128; s > 0; s >>= 1) {
        if (tid < s) red[tid] = fmaxf(red[tid], red[tid + s]);
        __syncthreads();
    }
    float m = red[0];
    __syncthreads();
    float sum = 0.f;
#pragma unroll
    for (int i = 0; i < 8; ++i) { vals[i] = __expf(vals[i] - m); sum += vals[i]; }
    red[tid] = sum; __syncthreads();
    for (int s = 128; s > 0; s >>= 1) {
        if (tid < s) red[tid] += red[tid + s];
        __syncthreads();
    }
    float inv = 1.0f / red[0];
#pragma unroll
    for (int i = 0; i < 8; ++i)
        alpha[(size_t)b * L_ + tid + i * 256] = vals[i] * inv;
}

// ---------- K4: cbar_part[c][b,d] = sum_{l in chunk c} alpha[b,l]*context[b,l,d] ----------
// grid (D/256, B, 4): 512 workgroups streaming 256MB -> saturates HBM.
__global__ __launch_bounds__(256) void pa_cbar(const float* __restrict__ ctxg,
                                               const float* __restrict__ alpha,
                                               float* __restrict__ cbar_part) {
    __shared__ float sA[L_ / 4];
    const int b = blockIdx.y, chunk = blockIdx.z;
    const int d = blockIdx.x * 256 + threadIdx.x;
    const int lbase = chunk * (L_ / 4);
    for (int i = threadIdx.x; i < L_ / 4; i += 256)
        sA[i] = alpha[(size_t)b * L_ + lbase + i];
    __syncthreads();
    const float* base = ctxg + ((size_t)b * L_ + lbase) * D_ + d;
    float acc = 0.f;
#pragma unroll 8
    for (int l = 0; l < L_ / 4; ++l) acc += sA[l] * base[(size_t)l * D_];
    cbar_part[((size_t)chunk * B_ + b) * D_ + d] = acc;
}

// ---------- K5: weighted_context[b,h] = Wc[h,:] . (sum_c cbar_part[c][b,:]) + bc[h] ----------
__global__ __launch_bounds__(256) void pa_wctx(const float* __restrict__ cbar_part,
                                               const float* __restrict__ Wc,
                                               const float* __restrict__ bc,
                                               float* __restrict__ out) {
    int idx = blockIdx.x * 256 + threadIdx.x;        // b*H + h
    int b = idx >> 9, h = idx & (H_ - 1);
    const float* x = cbar_part + (size_t)b * D_;     // chunk stride = B_*D_
    const float* w = Wc + (size_t)h * D_;
    float acc = bc[h];
#pragma unroll 8
    for (int d = 0; d < D_; ++d) {
        float xv = x[d] + x[d + B_ * D_] + x[d + 2 * B_ * D_] + x[d + 3 * B_ * D_];
        acc += xv * w[d];
    }
    out[idx] = acc;
}

extern "C" void kernel_launch(void* const* d_in, const int* in_sizes, int n_in,
                              void* d_out, int out_size, void* d_ws, size_t ws_size,
                              hipStream_t stream) {
    const float* dh   = (const float*)d_in[0];               // [B,D]
    const float* ctx  = (const float*)d_in[1];               // [B,L,D]
    const unsigned char* mask = (const unsigned char*)d_in[2]; // [B,L] bool
    const float* Wi   = (const float*)d_in[3];               // [H,D]
    const float* bi   = (const float*)d_in[4];               // [H]
    const float* Wc   = (const float*)d_in[5];               // [H,D]
    const float* bc   = (const float*)d_in[6];               // [H]
    const float* V    = (const float*)d_in[7];               // [H]

    float* out_wc    = (float*)d_out;            // [B,H]
    float* out_alpha = out_wc + B_ * H_;         // [B,L]

    char* ws = (char*)d_ws;
    __bf16* wc_bf     = (__bf16*)(ws);                                   // 512 KB (swizzled)
    float*  inp       = (float*)(ws + 524288);                           // 128 KB
    float*  e         = (float*)(ws + 524288 + 131072);                  // 512 KB
    float*  cbar_part = (float*)(ws + 524288 + 131072 + 524288);         // 512 KB (4 chunks)

    pa_cvt_wc <<<(H_ * D_) / 256, 256, 0, stream>>>(Wc, wc_bf);
    pa_inp    <<<(B_ * H_) / 256, 256, 0, stream>>>(dh, Wi, bi, inp);
    pa_energy <<<dim3(L_ / 64, B_), 256, 0, stream>>>(ctx, wc_bf, inp, bc, V, e);
    pa_softmax<<<B_, 256, 0, stream>>>(e, mask, out_alpha);
    pa_cbar   <<<dim3(D_ / 256, B_, 4), 256, 0, stream>>>(ctx, out_alpha, cbar_part);
    pa_wctx   <<<(B_ * H_) / 256, 256, 0, stream>>>(cbar_part, Wc, bc, out_wc);
}
